// GradStateSpaceBlock_36919538876644
// MI455X (gfx1250) — matrix operand order repaired and checked by reading, not compiled, not run on hardware
//
#include <hip/hip_runtime.h>
#include <hip/hip_bf16.h>
#include <math.h>

typedef _Float16 f16;
typedef __attribute__((ext_vector_type(16))) _Float16 v16h;
typedef __attribute__((ext_vector_type(8)))  float    v8f;

#define Bsz 8
#define Ntok 4096
#define DIM 256
#define HID 512
#define DTR 32
#define DS  16
#define Hdim 64
#define Wdim 64
#define Mrows (Bsz * Ntok)   // 32768

// ---------------------------------------------------------------------------
// Fragment-order ("swizzled") panel layouts, matching CDNA5 ISA 7.12.2 exactly.
// A panel: 16 rows x 32 k  = 512 halves; element (m_local, k_local) lives at
//   lane = m_local + 16*((k_local>>3)&1),  e = (k_local>>4)*8 + (k_local&7)
// B panel: 32 k x 16 cols = 512 halves; element (k_local, n_local) lives at
//   lane = n_local + 16*((k_local>>3)&1),  e = (k_local>>4)*8 + (k_local&7)
// Panel address = panel_index*512 + lane*16 + e  (each lane: contiguous 32B).
__device__ __forceinline__ size_t swA(int row, int c, int K) {
  int lane = (row & 15) + (((c >> 3) & 1) << 4);
  int e    = ((c >> 4) & 1) * 8 + (c & 7);
  return (((size_t)(row >> 4) * (K >> 5) + (c >> 5)) << 9) + lane * 16 + e;
}
__device__ __forceinline__ size_t swB(int k, int n, int Ktot) {
  int kl   = k & 31;
  int lane = (n & 15) + (((kl >> 3) & 1) << 4);
  int e    = ((kl >> 4) & 1) * 8 + (kl & 7);
  return (((size_t)(n >> 4) * (Ktot >> 5) + (k >> 5)) << 9) + lane * 16 + e;
}

// ---------------------------------------------------------------------------
// 1) gray = mean over channels
__global__ void k_gray(const float* __restrict__ x, float* __restrict__ gray) {
  int t = blockIdx.x;
  __shared__ float red[256];
  red[threadIdx.x] = x[(size_t)t * DIM + threadIdx.x];
  __syncthreads();
  for (int o = 128; o > 0; o >>= 1) {
    if (threadIdx.x < o) red[threadIdx.x] += red[threadIdx.x + o];
    __syncthreads();
  }
  if (threadIdx.x == 0) gray[t] = red[0] * (1.0f / DIM);
}

// 2) sobel/laplacian + fusion + abs
__global__ void k_gm(const float* __restrict__ gray, const float* __restrict__ fw,
                     float* __restrict__ gm) {
  int g = blockIdx.x * 256 + threadIdx.x;
  if (g >= Bsz * Ntok) return;
  int b = g >> 12, n = g & (Ntok - 1), h = n >> 6, w = n & 63;
  const float* G = gray + b * Ntok;
  float p[3][3];
#pragma unroll
  for (int dy = -1; dy <= 1; ++dy)
#pragma unroll
    for (int dx = -1; dx <= 1; ++dx) {
      int hh = h + dy, ww = w + dx;
      p[dy + 1][dx + 1] = (hh >= 0 && hh < Hdim && ww >= 0 && ww < Wdim)
                              ? G[hh * Wdim + ww] : 0.0f;
    }
  float sx = -p[0][0] + p[0][2] - 2.f * p[1][0] + 2.f * p[1][2] - p[2][0] + p[2][2];
  float sy = -p[0][0] - 2.f * p[0][1] - p[0][2] + p[2][0] + 2.f * p[2][1] + p[2][2];
  float lp = 4.f * p[1][1] - p[0][1] - p[1][0] - p[1][2] - p[2][1];
  gm[g] = fabsf(fw[0] * sx + fw[1] * sy + fw[2] * lp);
}

// 3) per-batch min/max normalize -> sigmoid priority
__global__ void k_gp(const float* __restrict__ gm, const float* __restrict__ ps,
                     const float* __restrict__ po, float* __restrict__ gp) {
  int b = blockIdx.x, tid = threadIdx.x;
  __shared__ float smin[1024], smax[1024];
  const float* G = gm + b * Ntok;
  float mn = 3.4e38f, mx = -3.4e38f;
  for (int i = tid; i < Ntok; i += 1024) {
    float v = G[i]; mn = fminf(mn, v); mx = fmaxf(mx, v);
  }
  smin[tid] = mn; smax[tid] = mx; __syncthreads();
  for (int o = 512; o > 0; o >>= 1) {
    if (tid < o) {
      smin[tid] = fminf(smin[tid], smin[tid + o]);
      smax[tid] = fmaxf(smax[tid], smax[tid + o]);
    }
    __syncthreads();
  }
  mn = smin[0]; mx = smax[0];
  float sc = ps[0], of = po[0], inv = 1.0f / (mx - mn + 1e-8f);
  for (int i = tid; i < Ntok; i += 1024) {
    float z = sc * ((G[i] - mn) * inv) + of;
    gp[b * Ntok + i] = 1.0f / (1.0f + expf(-z));
  }
}

// 4) stable-equivalent bitonic argsort per batch (key = (val, index))
__global__ void k_sort(const float* __restrict__ gp, int* __restrict__ idx,
                       int* __restrict__ inv) {
  int b = blockIdx.x, tid = threadIdx.x;
  __shared__ float sv[Ntok];
  __shared__ int   si[Ntok];
  for (int i = tid; i < Ntok; i += 1024) { sv[i] = gp[b * Ntok + i]; si[i] = i; }
  __syncthreads();
  for (int k = 2; k <= Ntok; k <<= 1)
    for (int j = k >> 1; j > 0; j >>= 1) {
      for (int t = tid; t < Ntok; t += 1024) {
        int ixj = t ^ j;
        if (ixj > t) {
          bool up = ((t & k) == 0);
          float v0 = sv[t], v1 = sv[ixj]; int i0 = si[t], i1 = si[ixj];
          bool gt = (v0 > v1) || (v0 == v1 && i0 > i1);
          if (gt == up) { sv[t] = v1; sv[ixj] = v0; si[t] = i1; si[ixj] = i0; }
        }
      }
      __syncthreads();
    }
  for (int i = tid; i < Ntok; i += 1024) {
    int s = si[i];
    idx[b * Ntok + i] = s;
    inv[b * Ntok + s] = i;
  }
}

// 5) generic LayerNorm -> f16 rows in swizzled A-panel layout
__global__ void k_ln_f16(const float* __restrict__ x, const float* __restrict__ g,
                         const float* __restrict__ bt, f16* __restrict__ out, int C) {
  int row = blockIdx.x, tid = threadIdx.x;
  __shared__ float red[256];
  const float* xr = x + (size_t)row * C;
  float s = 0.f;
  for (int c = tid; c < C; c += 256) s += xr[c];
  red[tid] = s; __syncthreads();
  for (int o = 128; o > 0; o >>= 1) { if (tid < o) red[tid] += red[tid + o]; __syncthreads(); }
  float mean = red[0] / C; __syncthreads();
  float v = 0.f;
  for (int c = tid; c < C; c += 256) { float d = xr[c] - mean; v += d * d; }
  red[tid] = v; __syncthreads();
  for (int o = 128; o > 0; o >>= 1) { if (tid < o) red[tid] += red[tid + o]; __syncthreads(); }
  float rstd = rsqrtf(red[0] / C + 1e-5f);
  for (int c = tid; c < C; c += 256)
    out[swA(row, c, C)] = (f16)((xr[c] - mean) * rstd * g[c] + bt[c]);
}

// 6) weight [O,I] f32 -> swizzled B panels (GEMM sees B[k=i][n=o])
__global__ void k_wswz(const float* __restrict__ w, f16* __restrict__ out, int O, int I) {
  int g = blockIdx.x * 256 + threadIdx.x;
  if (g < O * I) { int o = g / I, i = g % I; out[swB(i, o, I)] = (f16)w[g]; }
}

// 7) WMMA GEMM on swizzled panels, no LDS.  Wave tile: 32x32 (4 WMMA tiles,
//    4 fragment loads per K-step -> 1.0 loads/WMMA).  Block: 8 waves = 128x64.
//    out[M,N] = act(A_sw[M,K] * B_sw[K,N] + bias) (+residual)
//    act: 0=none 1=silu 2=softplus.  Writes f32 (row-major) and/or f16 (swizzled).
__global__ void k_gemm_sw(const f16* __restrict__ A, const f16* __restrict__ Bw,
                          const float* __restrict__ bias,
                          float* __restrict__ outF, f16* __restrict__ outH,
                          const float* __restrict__ residual,
                          int M, int N, int K, int act) {
  const int lane = threadIdx.x & 31, wv = threadIdx.x >> 5;
  const int wm = wv >> 1, wn = wv & 1;           // 4 x 32 rows, 2 x 32 cols
  const int gm0 = blockIdx.y * 128, gn0 = blockIdx.x * 64;
  const int Kb = K >> 5;
  const int mblk0 = (gm0 >> 4) + wm * 2;
  const int nblk0 = (gn0 >> 4) + wn * 2;

  const f16* Ap0 = A  + ((size_t)mblk0 * Kb << 9) + lane * 16;
  const f16* Ap1 = A  + ((size_t)(mblk0 + 1) * Kb << 9) + lane * 16;
  const f16* Bp0 = Bw + ((size_t)nblk0 * Kb << 9) + lane * 16;
  const f16* Bp1 = Bw + ((size_t)(nblk0 + 1) * Kb << 9) + lane * 16;

  v8f acc00 = {}, acc01 = {}, acc10 = {}, acc11 = {};
#pragma unroll 2
  for (int kb = 0; kb < Kb; ++kb) {
    size_t o = (size_t)kb << 9;
    v16h a0 = *(const v16h*)(Ap0 + o);
    v16h a1 = *(const v16h*)(Ap1 + o);
    v16h b0 = *(const v16h*)(Bp0 + o);
    v16h b1 = *(const v16h*)(Bp1 + o);
    acc00 = __builtin_amdgcn_wmma_f32_16x16x32_f16(false, a0, false, b0, (short)0, acc00, false, false);
    acc01 = __builtin_amdgcn_wmma_f32_16x16x32_f16(false, a0, false, b1, (short)0, acc01, false, false);
    acc10 = __builtin_amdgcn_wmma_f32_16x16x32_f16(false, a1, false, b0, (short)0, acc10, false, false);
    acc11 = __builtin_amdgcn_wmma_f32_16x16x32_f16(false, a1, false, b1, (short)0, acc11, false, false);
  }

  int n = lane & 15, mh = (lane >> 4) * 8;
#pragma unroll
  for (int i = 0; i < 2; ++i) {          // m sub-tile
#pragma unroll
    for (int j = 0; j < 2; ++j) {        // n sub-tile
      v8f acc = (i == 0) ? ((j == 0) ? acc00 : acc01)
                         : ((j == 0) ? acc10 : acc11);
#pragma unroll
      for (int v = 0; v < 8; ++v) {
        int row = gm0 + wm * 32 + i * 16 + mh + v;
        int col = gn0 + wn * 32 + j * 16 + n;
        float val = acc[v];
        if (bias) val += bias[col];
        if (act == 1) val = val * (1.0f / (1.0f + expf(-val)));           // silu
        else if (act == 2) val = (val > 20.f) ? val : log1pf(expf(val));  // softplus
        if (residual) val += residual[(size_t)row * N + col];
        if (outF) outF[(size_t)row * N + col] = val;
        if (outH) outH[swA(row, col, N)] = (f16)val;
      }
    }
  }
}

// 8) depthwise 3x3 CPE + sigmoid gate: tok = xp * sigmoid(dwconv(xp)+b)
__global__ void k_cpe(const float* __restrict__ xp, const float* __restrict__ cw,
                      const float* __restrict__ cb, float* __restrict__ tok) {
  size_t g = (size_t)blockIdx.x * 256 + threadIdx.x;
  int d = (int)(g & (HID - 1));
  size_t bn = g >> 9;
  int n = (int)(bn & (Ntok - 1)), b = (int)(bn >> 12);
  int h = n >> 6, w = n & 63;
  const float* X = xp + ((size_t)b * Ntok) * HID;
  float acc = cb[d];
#pragma unroll
  for (int dy = -1; dy <= 1; ++dy) {
    int hh = h + dy; if (hh < 0 || hh >= Hdim) continue;
#pragma unroll
    for (int dx = -1; dx <= 1; ++dx) {
      int ww = w + dx; if (ww < 0 || ww >= Wdim) continue;
      acc += cw[d * 9 + (dy + 1) * 3 + (dx + 1)] * X[(size_t)(hh * Wdim + ww) * HID + d];
    }
  }
  float xv = X[(size_t)n * HID + d];
  tok[g] = xv * (1.0f / (1.0f + expf(-acc)));
}

// 9) gather tokens by priority order (f32 row-major for scan, f16 swizzled for GEMM)
__global__ void k_gather(const float* __restrict__ tok, const int* __restrict__ idx,
                         float* __restrict__ sxf, f16* __restrict__ sxh) {
  size_t g = (size_t)blockIdx.x * 256 + threadIdx.x;
  int d = (int)(g & (HID - 1));
  size_t bl = g >> 9;
  int l = (int)(bl & (Ntok - 1)), b = (int)(bl >> 12);
  int src = idx[b * Ntok + l];
  float v = tok[(((size_t)b * Ntok + src) << 9) + d];
  sxf[g] = v;
  sxh[swA((int)bl, d, HID)] = (f16)v;
}

// 10) split x_dbl: dts -> swizzled f16; C-correction term (faithful .view reinterpret)
__global__ void k_split(const float* __restrict__ xdbl, const float* __restrict__ gp,
                        const float* __restrict__ gCw, const float* __restrict__ gCb,
                        const float* __restrict__ gCweight,
                        f16* __restrict__ dts, float* __restrict__ cadd) {
  size_t g = (size_t)blockIdx.x * 256 + threadIdx.x;
  size_t total_d = (size_t)Mrows * DTR;
  if (g < total_d) {
    size_t m = g >> 5; int r = (int)(g & 31);
    dts[swA((int)m, r, DTR)] = (f16)xdbl[m * 64 + r];
  }
  size_t total_c = (size_t)Bsz * DS * Ntok;
  if (g < total_c) {
    int l = (int)(g & (Ntok - 1));
    size_t bs = g >> 12;
    int s = (int)(bs & 15), b = (int)(bs >> 4);
    float gw = gCweight[0];
    cadd[g] = gw * (gp[b * Ntok + s * 256 + (l >> 4)] * gCw[l & 15] + gCb[l & 15]);
  }
}

// 11) selective scan: one thread per (b,d); y overwrites delta buffer in place
__global__ void k_scan(const float* __restrict__ u, float* __restrict__ dy,
                       const float* __restrict__ xdbl, const float* __restrict__ cadd,
                       const float* __restrict__ A_logs, const float* __restrict__ Ds) {
  int t = blockIdx.x * 256 + threadIdx.x;   // B*HID = 4096
  int b = t >> 9, d = t & (HID - 1);
  float A[DS], h[DS];
#pragma unroll
  for (int s = 0; s < DS; ++s) { A[s] = -expf(A_logs[d * DS + s]); h[s] = 0.0f; }
  float Dd = Ds[d];
  const float* Xb = xdbl + (size_t)b * Ntok * 64;
  const float* Cb = cadd + (size_t)b * DS * Ntok;
  size_t base = ((size_t)b * Ntok) * HID + d;
  for (int l = 0; l < Ntok; ++l) {
    size_t o = base + (size_t)l * HID;
    float dt = dy[o];
    float uv = u[o];
    const float* row = Xb + (size_t)l * 64;
    float du = dt * uv, y = 0.0f;
#pragma unroll
    for (int s = 0; s < DS; ++s) {
      float hv = expf(dt * A[s]) * h[s] + du * row[32 + s];
      h[s] = hv;
      y += hv * (row[48 + s] + Cb[s * Ntok + l]);
    }
    dy[o] = y + uv * Dd;
  }
}

// 12) un-sort + residual: x1 = x + xo[inv]
__global__ void k_unsort_add(const float* __restrict__ x, const float* __restrict__ xo,
                             const int* __restrict__ inv, float* __restrict__ x1) {
  size_t g = (size_t)blockIdx.x * 256 + threadIdx.x;
  int c = (int)(g & (DIM - 1));
  size_t bn = g >> 8;
  int n = (int)(bn & (Ntok - 1)), b = (int)(bn >> 12);
  int src = inv[b * Ntok + n];
  x1[g] = x[g] + xo[(((size_t)b * Ntok + src) << 8) + c];
}

// ---------------------------------------------------------------------------
extern "C" void kernel_launch(void* const* d_in, const int* in_sizes, int n_in,
                              void* d_out, int out_size, void* d_ws, size_t ws_size,
                              hipStream_t stream) {
  const float* x        = (const float*)d_in[0];
  const float* norm1_g  = (const float*)d_in[1];
  const float* norm1_b  = (const float*)d_in[2];
  const float* fusion_w = (const float*)d_in[3];
  const float* prio_sc  = (const float*)d_in[4];
  const float* prio_of  = (const float*)d_in[5];
  const float* in_w     = (const float*)d_in[6];
  const float* in_b     = (const float*)d_in[7];
  const float* cpe_w    = (const float*)d_in[8];
  const float* cpe_b    = (const float*)d_in[9];
  const float* x_proj_w = (const float*)d_in[10];
  const float* dt_w     = (const float*)d_in[11];
  const float* dt_b     = (const float*)d_in[12];
  const float* A_logs   = (const float*)d_in[13];
  const float* Ds       = (const float*)d_in[14];
  const float* gC_w     = (const float*)d_in[15];
  const float* gC_b     = (const float*)d_in[16];
  const float* gC_wt    = (const float*)d_in[17];
  const float* outn_g   = (const float*)d_in[18];
  const float* outn_b   = (const float*)d_in[19];
  const float* outp_w   = (const float*)d_in[20];
  const float* outp_b   = (const float*)d_in[21];
  const float* norm2_g  = (const float*)d_in[22];
  const float* norm2_b  = (const float*)d_in[23];
  const float* mlp_w1   = (const float*)d_in[24];
  const float* mlp_b1   = (const float*)d_in[25];
  const float* mlp_w2   = (const float*)d_in[26];
  const float* mlp_b2   = (const float*)d_in[27];
  float* out = (float*)d_out;

  // ---- workspace carve-up ----
  char* base = (char*)d_ws;
  size_t off = 0;
  auto take = [&](size_t bytes) {
    size_t o = off;
    off = (off + bytes + 255) & ~(size_t)255;
    return (void*)(base + o);
  };
  float* gray  = (float*)take((size_t)Mrows * 4);
  float* gmb   = (float*)take((size_t)Mrows * 4);
  float* gp    = (float*)take((size_t)Mrows * 4);
  int*   idx   = (int*)  take((size_t)Mrows * 4);
  int*   inv   = (int*)  take((size_t)Mrows * 4);
  f16*   wIN   = (f16*)  take((size_t)DIM * HID * 2);
  f16*   wXP   = (f16*)  take((size_t)HID * 64 * 2);
  f16*   wDT   = (f16*)  take((size_t)DTR * HID * 2);
  f16*   wOP   = (f16*)  take((size_t)HID * DIM * 2);
  f16*   wM1   = (f16*)  take((size_t)DIM * HID * 2);
  f16*   wM2   = (f16*)  take((size_t)HID * DIM * 2);
  f16*   xn16  = (f16*)  take((size_t)Mrows * DIM * 2);
  f16*   h16   = (f16*)  take((size_t)Mrows * HID * 2);
  float* S1    = (float*)take((size_t)Mrows * HID * 4); // xp -> delta/y (aliased)
  float* S2    = (float*)take((size_t)Mrows * HID * 4); // tok
  float* S3    = (float*)take((size_t)Mrows * HID * 4); // sx f32
  float* xdbl  = (float*)take((size_t)Mrows * 64 * 4);
  f16*   dts16 = (f16*)  take((size_t)Mrows * DTR * 2);
  float* cadd  = (float*)take((size_t)Bsz * DS * Ntok * 4);
  float* xo    = (float*)take((size_t)Mrows * DIM * 4);
  float* x1    = (float*)take((size_t)Mrows * DIM * 4);

  // ---- priority path ----
  k_gray<<<Mrows, 256, 0, stream>>>(x, gray);
  k_gm<<<(Mrows + 255) / 256, 256, 0, stream>>>(gray, fusion_w, gmb);
  k_gp<<<Bsz, 1024, 0, stream>>>(gmb, prio_sc, prio_of, gp);
  k_sort<<<Bsz, 1024, 0, stream>>>(gp, idx, inv);

  // ---- weight conversions (f32 -> swizzled f16 B panels) ----
  k_wswz<<<(HID * DIM + 255) / 256, 256, 0, stream>>>(in_w,     wIN, HID, DIM);
  k_wswz<<<(64 * HID + 255) / 256,  256, 0, stream>>>(x_proj_w, wXP, 64,  HID);
  k_wswz<<<(HID * DTR + 255) / 256, 256, 0, stream>>>(dt_w,     wDT, HID, DTR);
  k_wswz<<<(DIM * HID + 255) / 256, 256, 0, stream>>>(outp_w,   wOP, DIM, HID);
  k_wswz<<<(HID * DIM + 255) / 256, 256, 0, stream>>>(mlp_w1,   wM1, HID, DIM);
  k_wswz<<<(DIM * HID + 255) / 256, 256, 0, stream>>>(mlp_w2,   wM2, DIM, HID);

  // ---- mamba branch ----
  k_ln_f16<<<Mrows, 256, 0, stream>>>(x, norm1_g, norm1_b, xn16, DIM);
  k_gemm_sw<<<dim3(HID / 64, Mrows / 128), 256, 0, stream>>>(
      xn16, wIN, in_b, S1, nullptr, nullptr, Mrows, HID, DIM, 0);          // xp
  k_cpe<<<(int)(((size_t)Mrows * HID) / 256), 256, 0, stream>>>(S1, cpe_w, cpe_b, S2);
  k_gather<<<(int)(((size_t)Mrows * HID) / 256), 256, 0, stream>>>(S2, idx, S3, h16);
  k_gemm_sw<<<dim3(64 / 64, Mrows / 128), 256, 0, stream>>>(
      h16, wXP, nullptr, xdbl, nullptr, nullptr, Mrows, 64, HID, 0);       // x_dbl
  k_split<<<(int)(((size_t)Mrows * DTR) / 256), 256, 0, stream>>>(
      xdbl, gp, gC_w, gC_b, gC_wt, dts16, cadd);
  k_gemm_sw<<<dim3(HID / 64, Mrows / 128), 256, 0, stream>>>(
      dts16, wDT, dt_b, S1, nullptr, nullptr, Mrows, HID, DTR, 2);         // delta
  k_scan<<<(Bsz * HID) / 256, 256, 0, stream>>>(S3, S1, xdbl, cadd, A_logs, Ds);
  k_ln_f16<<<Mrows, 256, 0, stream>>>(S1, outn_g, outn_b, h16, HID);
  k_gemm_sw<<<dim3(DIM / 64, Mrows / 128), 256, 0, stream>>>(
      h16, wOP, outp_b, xo, nullptr, nullptr, Mrows, DIM, HID, 0);         // out_proj
  k_unsort_add<<<(int)(((size_t)Mrows * DIM) / 256), 256, 0, stream>>>(x, xo, inv, x1);

  // ---- MLP branch ----
  k_ln_f16<<<Mrows, 256, 0, stream>>>(x1, norm2_g, norm2_b, xn16, DIM);
  k_gemm_sw<<<dim3(HID / 64, Mrows / 128), 256, 0, stream>>>(
      xn16, wM1, mlp_b1, nullptr, h16, nullptr, Mrows, HID, DIM, 1);       // silu
  k_gemm_sw<<<dim3(DIM / 64, Mrows / 128), 256, 0, stream>>>(
      h16, wM2, mlp_b2, out, nullptr, x1, Mrows, DIM, HID, 0);             // + x1
}
